// MiniMaxM2Attention_88579405513103
// MI455X (gfx1250) — matrix operand order repaired and checked by reading, not compiled, not run on hardware
//
#include <hip/hip_runtime.h>
#include <hip/hip_bf16.h>

// ---------------------------------------------------------------------------
// MiniMax-M2 attention block for MI455X (gfx1250, wave32, WMMA).
// Pipeline: QKV GEMM (bf16 WMMA) -> RMSNorm+RoPE -> flash attention (WMMA,
//           async-to-LDS K staging) -> output projection GEMM (bf16 WMMA).
// ---------------------------------------------------------------------------

typedef __attribute__((ext_vector_type(16))) __bf16 v16bf;
typedef __attribute__((ext_vector_type(8)))  __bf16 v8bf;
typedef __attribute__((ext_vector_type(4)))  __bf16 v4bf;
typedef __attribute__((ext_vector_type(8)))  float  v8f;

constexpr int kB   = 2;
constexpr int kS   = 2048;
constexpr int kH   = 2048;
constexpr int kNH  = 16;
constexpr int kNKV = 4;
constexpr int kGRP = kNH / kNKV;      // 4
constexpr int kHD  = 128;
constexpr int kRD  = 64;
constexpr int kWIN = 1024;
constexpr int kSP  = kS + 32;         // padded V-transposed row length
constexpr float kEPS   = 1e-6f;
constexpr float kSCALE = 0.08838834764831845f;  // 128^-0.5
constexpr float kTHETA = 1.0e6f;

__device__ __forceinline__ v8f wmma_bf16(v16bf a, v16bf b, v8f c) {
  return __builtin_amdgcn_wmma_f32_16x16x32_bf16(
      /*neg_a=*/false, a, /*neg_b=*/false, b,
      /*c_mod=*/(short)0, c, /*reuse_a=*/false, /*reuse_b=*/false);
}

__device__ __forceinline__ v16bf cat8(v8bf lo, v8bf hi) {
  return __builtin_shufflevector(lo, hi, 0,1,2,3,4,5,6,7,8,9,10,11,12,13,14,15);
}

// ---------------------------------------------------------------------------
// Kernel 1/4: generic f32 GEMM via bf16 WMMA.  C[M,N] = A[M,K] * B[K,N].
// Block: 256 threads (8 waves).  Tile: BM=128, BN=128, BK=32.
// Wave (wm=wave&3, wn=wave>>2) owns a 32x64 macro tile = 2x4 WMMA frags.
// ---------------------------------------------------------------------------
#define BM 128
#define BN 128
#define BK 32

__global__ __launch_bounds__(256) void gemm_bf16_wmma_kernel(
    const float* __restrict__ A, const float* __restrict__ B,
    float* __restrict__ C, int M, int N, int K) {
  __shared__ __align__(32) __bf16 sA[BM * BK];   // [m][k]
  __shared__ __align__(32) __bf16 sB[BN * BK];   // transposed: [n][k]

  const int tid    = threadIdx.x;
  const int lane   = tid & 31;
  const int wave   = tid >> 5;
  const int lanelo = lane & 15;
  const int ha     = lane >> 4;
  const int wm     = wave & 3;
  const int wn     = wave >> 2;
  const int m0     = blockIdx.y * BM;
  const int n0     = blockIdx.x * BN;

  // staging coordinates: A -> 16 contiguous f32 per thread along K
  const int am = tid >> 1;             // 0..127
  const int ak = (tid & 1) * 16;       // 0 or 16
  // B -> 4(k) x 4(n) micro-block per thread, stored transposed as b64 packs
  const int bn4 = (tid & 31) * 4;      // n offset: 0..124
  const int bkq = (tid >> 5) * 4;      // k offset: 0..28

  v8f acc[2][4];
  const v8f vzero = {0.f, 0.f, 0.f, 0.f, 0.f, 0.f, 0.f, 0.f};
#pragma unroll
  for (int mi = 0; mi < 2; ++mi)
#pragma unroll
    for (int ni = 0; ni < 4; ++ni) acc[mi][ni] = vzero;

  for (int k0 = 0; k0 < K; k0 += BK) {
    const float* ap = A + (size_t)(m0 + am) * K + k0 + ak;
    const float* bp = B + (size_t)(k0 + bkq) * N + n0 + bn4;
#pragma unroll
    for (int i = 0; i < 16; i += 4) {
      float4 v = *(const float4*)(ap + i);
      sA[am * BK + ak + i + 0] = (__bf16)v.x;
      sA[am * BK + ak + i + 1] = (__bf16)v.y;
      sA[am * BK + ak + i + 2] = (__bf16)v.z;
      sA[am * BK + ak + i + 3] = (__bf16)v.w;
    }
    {
      float4 r0 = *(const float4*)(bp);
      float4 r1 = *(const float4*)(bp + N);
      float4 r2 = *(const float4*)(bp + 2 * (size_t)N);
      float4 r3 = *(const float4*)(bp + 3 * (size_t)N);
      v4bf c0 = {(__bf16)r0.x, (__bf16)r1.x, (__bf16)r2.x, (__bf16)r3.x};
      v4bf c1 = {(__bf16)r0.y, (__bf16)r1.y, (__bf16)r2.y, (__bf16)r3.y};
      v4bf c2 = {(__bf16)r0.z, (__bf16)r1.z, (__bf16)r2.z, (__bf16)r3.z};
      v4bf c3 = {(__bf16)r0.w, (__bf16)r1.w, (__bf16)r2.w, (__bf16)r3.w};
      *(v4bf*)(sB + (bn4 + 0) * BK + bkq) = c0;   // ds_store_b64
      *(v4bf*)(sB + (bn4 + 1) * BK + bkq) = c1;
      *(v4bf*)(sB + (bn4 + 2) * BK + bkq) = c2;
      *(v4bf*)(sB + (bn4 + 3) * BK + bkq) = c3;
    }
    if (k0 + BK < K) {                      // -> global_prefetch_b8
      __builtin_prefetch(ap + BK, 0, 1);
      __builtin_prefetch(bp + (size_t)BK * N, 0, 1);
    }
    __syncthreads();

    v16bf bF[4];
#pragma unroll
    for (int ni = 0; ni < 4; ++ni) {
      const int nc = wn * 64 + ni * 16 + lanelo;
      bF[ni] = *(const v16bf*)(sB + nc * BK + ha * 16);
    }
#pragma unroll
    for (int mi = 0; mi < 2; ++mi) {
      const int mr = wm * 32 + mi * 16 + lanelo;
      v8bf lo = *(const v8bf*)(sA + mr * BK + ha * 16);
      v8bf hi = *(const v8bf*)(sA + mr * BK + ha * 16 + 8);
      v16bf aF = cat8(lo, hi);
#pragma unroll
      for (int ni = 0; ni < 4; ++ni)
        acc[mi][ni] = wmma_bf16(aF, bF[ni], acc[mi][ni]);
    }
    __syncthreads();
  }

#pragma unroll
  for (int mi = 0; mi < 2; ++mi)
#pragma unroll
    for (int ni = 0; ni < 4; ++ni) {
      const int n = n0 + wn * 64 + ni * 16 + lanelo;
#pragma unroll
      for (int j = 0; j < 8; ++j) {
        const int m = m0 + wm * 32 + mi * 16 + j + 8 * ha;
        C[(size_t)m * N + n] = acc[mi][ni][j];
      }
    }
}

// ---------------------------------------------------------------------------
// Kernel 2: joint RMSNorm (over full 2048/512 dims) + partial RoPE.
// One block per token.  Emits:
//   Qh bf16 [b, h, s, 128]          (row-major rows for A/B WMMA frag loads)
//   Kh bf16 [b, kh, s, 128]
//   Vt bf16 [b, kh, 128, S+32]      (transposed so PV B-frags are contiguous)
// ---------------------------------------------------------------------------
__global__ __launch_bounds__(256) void norm_rope_kernel(
    const float* __restrict__ Qf, const float* __restrict__ Kf,
    const float* __restrict__ Vf, const float* __restrict__ qw,
    const float* __restrict__ kw, __bf16* __restrict__ Qh,
    __bf16* __restrict__ Kh, __bf16* __restrict__ Vt) {
  const int t = blockIdx.x;
  const int b = t / kS;
  const int s = t % kS;
  const int tid = threadIdx.x;
  const float pos = (float)s;   // position_ids == arange(S) per setup

  __shared__ float red[256];

  const float* qrow = Qf + (size_t)t * (kNH * kHD);
  const float* krow = Kf + (size_t)t * (kNKV * kHD);
  const float* vrow = Vf + (size_t)t * (kNKV * kHD);

  // --- RMS of q row (2048 elems) ---
  float ss = 0.f;
#pragma unroll
  for (int i = 0; i < 8; ++i) {
    float x = qrow[tid + i * 256];
    ss += x * x;
  }
  red[tid] = ss;
  __syncthreads();
  for (int off = 128; off > 0; off >>= 1) {
    if (tid < off) red[tid] += red[tid + off];
    __syncthreads();
  }
  const float rq = rsqrtf(red[0] / (float)(kNH * kHD) + kEPS);
  __syncthreads();

  // --- RMS of k row (512 elems) ---
  float x0 = krow[tid], x1 = krow[tid + 256];
  red[tid] = x0 * x0 + x1 * x1;
  __syncthreads();
  for (int off = 128; off > 0; off >>= 1) {
    if (tid < off) red[tid] += red[tid + off];
    __syncthreads();
  }
  const float rk = rsqrtf(red[0] / (float)(kNKV * kHD) + kEPS);
  __syncthreads();

  // --- Q: norm + rope, write [b,h,s,hd] bf16 ---
#pragma unroll
  for (int i = 0; i < 8; ++i) {
    const int d = tid + i * 256;
    const int head = d >> 7;
    const int hd = d & 127;
    float x = qrow[d] * rq * qw[d];
    float val = x;
    if (hd < kRD) {
      const int j = hd & 31;
      const float th = pos * __powf(kTHETA, -(float)(2 * j) / (float)kRD);
      const float c = cosf(th), sn = sinf(th);
      if (hd < kRD / 2) {
        const float xp = qrow[d + 32] * rq * qw[d + 32];
        val = x * c - xp * sn;
      } else {
        const float xp = qrow[d - 32] * rq * qw[d - 32];
        val = x * c + xp * sn;
      }
    }
    Qh[((size_t)(b * kNH + head) * kS + s) * kHD + hd] = (__bf16)val;
  }

  // --- K: norm + rope, V: transpose ---
#pragma unroll
  for (int i = 0; i < 2; ++i) {
    const int d = tid + i * 256;
    const int head = d >> 7;
    const int hd = d & 127;
    float x = krow[d] * rk * kw[d];
    float val = x;
    if (hd < kRD) {
      const int j = hd & 31;
      const float th = pos * __powf(kTHETA, -(float)(2 * j) / (float)kRD);
      const float c = cosf(th), sn = sinf(th);
      if (hd < kRD / 2) {
        const float xp = krow[d + 32] * rk * kw[d + 32];
        val = x * c - xp * sn;
      } else {
        const float xp = krow[d - 32] * rk * kw[d - 32];
        val = x * c + xp * sn;
      }
    }
    Kh[((size_t)(b * kNKV + head) * kS + s) * kHD + hd] = (__bf16)val;
    Vt[((size_t)(b * kNKV + head) * kHD + hd) * kSP + s] = (__bf16)vrow[d];
  }
}

// ---------------------------------------------------------------------------
// Kernel 3: sliding-window causal flash attention, one wave per 16-row q tile.
// 32 keys per iteration.  The 8KB K-tile is staged into LDS with CDNA5
// GLOBAL_LOAD_ASYNC_TO_LDS_B128 (double-buffered, ASYNCcnt-synchronized) so
// the DMA of tile i+1 overlaps the WMMAs of tile i without burning VGPRs.
// Two 16x16 score WMMAs, online softmax via wave shuffles, P transposed
// through LDS (s_wait_dscnt fence), then 8 PV WMMAs with K=32 contraction.
// Output AO f32 [b*S, NH*HD] (token-major, feeds the out-proj GEMM).
// ---------------------------------------------------------------------------
__device__ __forceinline__ void async_stage_k(const __bf16* kbase, int kt,
                                              unsigned ldsbase, int lane) {
  const int rhalf = lane >> 4;         // which of 2 rows this lane serves
  const int seg   = lane & 15;         // 16-byte segment within a 256B row
#pragma unroll
  for (int i = 0; i < 16; ++i) {       // 16 issues x 512B = 8KB tile
    int row = kt + 2 * i + rhalf;
    if (row > kS - 1) row = kS - 1;    // clamped rows are masked later
    unsigned long long g =
        (unsigned long long)(uintptr_t)(kbase + (size_t)row * kHD + seg * 8);
    unsigned l = ldsbase + (unsigned)(((2 * i + rhalf) * kHD + seg * 8) * 2);
    asm volatile("global_load_async_to_lds_b128 %0, %1, off"
                 :: "v"(l), "v"(g) : "memory");
  }
}

__global__ __launch_bounds__(128) void flash_wmma_kernel(
    const __bf16* __restrict__ Qh, const __bf16* __restrict__ Kh,
    const __bf16* __restrict__ Vt, float* __restrict__ AO) {
  const int lane   = threadIdx.x & 31;
  const int wave   = threadIdx.x >> 5;
  const int lanelo = lane & 15;
  const int ha     = lane >> 4;

  const int QT = kS / 16;                       // 128 q-tiles per head
  const int task = blockIdx.x * 4 + wave;
  const int qt = task % QT;
  const int h  = (task / QT) % kNH;
  const int b  = task / (QT * kNH);
  const int q0 = qt * 16;
  const int kh = h / kGRP;

  __shared__ __align__(32) __bf16 sK[4][2][32 * kHD];  // 2x8KB per wave
  __shared__ __align__(32) __bf16 sP[4][16 * 32];
  __bf16* myP = sP[wave];
  const unsigned ldsK = (unsigned)(uintptr_t)&sK[wave][0][0];

  // Q fragments: 16x128 = 4 chunks of 16x32 (A-matrix layout)
  v16bf aQ[4];
  {
    const __bf16* qrow =
        Qh + ((size_t)(b * kNH + h) * kS + (q0 + lanelo)) * kHD;
#pragma unroll
    for (int c = 0; c < 4; ++c) {
      v8bf lo = *(const v8bf*)(qrow + c * 32 + ha * 16);
      v8bf hi = *(const v8bf*)(qrow + c * 32 + ha * 16 + 8);
      aQ[c] = cat8(lo, hi);
    }
  }

  const v8f vzero = {0.f, 0.f, 0.f, 0.f, 0.f, 0.f, 0.f, 0.f};
  v8f accO[8];
#pragma unroll
  for (int f = 0; f < 8; ++f) accO[f] = vzero;
  float mrow[8], lrow[8];
#pragma unroll
  for (int j = 0; j < 8; ++j) { mrow[j] = -1e30f; lrow[j] = 0.f; }

  const __bf16* kbase = Kh + (size_t)(b * kNKV + kh) * kS * kHD;
  const __bf16* vbase = Vt + (size_t)(b * kNKV + kh) * kHD * kSP;

  int kstart = q0 - kWIN;
  if (kstart < 0) kstart = 0;
  kstart &= ~31;

  // prime the async pipeline with the first K tile
  async_stage_k(kbase, kstart, ldsK, lane);
  int p = 0;

  for (int kt = kstart; kt <= q0; kt += 32) {
    // wait until buffer p holds tile kt, then kick off the DMA for kt+32
    asm volatile("s_wait_asynccnt 0x0" ::: "memory");
    if (kt + 32 <= q0)
      async_stage_k(kbase, kt + 32, ldsK + (unsigned)((1 - p) * 32 * kHD * 2),
                    lane);
    const __bf16* myK = &sK[wave][p][0];

    // ---- scores: 16x32 tile = two 16x16 WMMA accumulators (K from LDS) ----
    v8f s0 = vzero, s1 = vzero;
    {
      const __bf16* kr0 = myK + (size_t)lanelo * kHD;
      const __bf16* kr1 = myK + (size_t)(16 + lanelo) * kHD;
#pragma unroll
      for (int c = 0; c < 4; ++c) {
        v16bf b0 = *(const v16bf*)(kr0 + c * 32 + ha * 16);
        v16bf b1 = *(const v16bf*)(kr1 + c * 32 + ha * 16);
        s0 = wmma_bf16(aQ[c], b0, s0);
        s1 = wmma_bf16(aQ[c], b1, s1);
      }
    }
    // ---- mask + online softmax (per row, replicated across 16-lane half) --
#pragma unroll
    for (int j = 0; j < 8; ++j) {
      const int M = q0 + j + 8 * ha;
      const int N0 = kt + lanelo;
      const int N1 = N0 + 16;
      const bool ok0 = (N0 <= M) && (N0 + kWIN >= M);
      const bool ok1 = (N1 <= M) && (N1 + kWIN >= M);
      float sc0 = ok0 ? s0[j] * kSCALE : -1e30f;
      float sc1 = ok1 ? s1[j] * kSCALE : -1e30f;
      float mx = fmaxf(sc0, sc1);
#pragma unroll
      for (int off = 1; off < 16; off <<= 1) mx = fmaxf(mx, __shfl_xor(mx, off));
      const float newm = fmaxf(mrow[j], mx);
      const float r = __expf(mrow[j] - newm);
      const float p0 = ok0 ? __expf(sc0 - newm) : 0.f;
      const float p1 = ok1 ? __expf(sc1 - newm) : 0.f;
      float ps = p0 + p1;
#pragma unroll
      for (int off = 1; off < 16; off <<= 1) ps += __shfl_xor(ps, off);
      lrow[j] = lrow[j] * r + ps;
      mrow[j] = newm;
#pragma unroll
      for (int f = 0; f < 8; ++f) accO[f][j] *= r;
      // store P tile to LDS in natural [row][col] order (C -> A relayout)
      myP[(j + 8 * ha) * 32 + lanelo] = (__bf16)p0;
      myP[(j + 8 * ha) * 32 + 16 + lanelo] = (__bf16)p1;
    }
    // same-wave LDS RAW fence (cross-lane dependence the compiler can't see)
    asm volatile("s_wait_dscnt 0x0" ::: "memory");

    // ---- reload P as 16x32 A-matrix fragment ----
    v8bf plo = *(const v8bf*)(myP + lanelo * 32 + ha * 16);
    v8bf phi = *(const v8bf*)(myP + lanelo * 32 + ha * 16 + 8);
    v16bf aP = cat8(plo, phi);

    // ---- PV: B = V^T chunk 32x128 -> 8 WMMAs (contiguous frag loads) ----
#pragma unroll
    for (int f = 0; f < 8; ++f) {
      v16bf bV = *(const v16bf*)(vbase + (size_t)(f * 16 + lanelo) * kSP +
                                 kt + ha * 16);
      accO[f] = wmma_bf16(aP, bV, accO[f]);
    }
    p ^= 1;
  }

  // ---- epilogue: normalize and store token-major [b*S, NH*HD] ----
#pragma unroll
  for (int j = 0; j < 8; ++j) {
    const int srow = q0 + j + 8 * ha;
    const float inv = 1.f / lrow[j];
#pragma unroll
    for (int f = 0; f < 8; ++f) {
      const int col = h * kHD + f * 16 + lanelo;
      AO[(size_t)(b * kS + srow) * (kNH * kHD) + col] = accO[f][j] * inv;
    }
  }
}

// ---------------------------------------------------------------------------
// Launch
// ---------------------------------------------------------------------------
extern "C" void kernel_launch(void* const* d_in, const int* in_sizes, int n_in,
                              void* d_out, int out_size, void* d_ws,
                              size_t ws_size, hipStream_t stream) {
  (void)in_sizes; (void)n_in; (void)out_size; (void)ws_size;

  const float* X  = (const float*)d_in[0];   // hidden_states [B,S,H] f32
  // d_in[1] position_ids: equals broadcast arange(S); positions derived from s
  const float* Wq = (const float*)d_in[2];
  const float* Wk = (const float*)d_in[3];
  const float* Wv = (const float*)d_in[4];
  const float* Wo = (const float*)d_in[5];
  const float* qw = (const float*)d_in[6];
  const float* kw = (const float*)d_in[7];
  float* out = (float*)d_out;

  const int M = kB * kS;                     // 4096 tokens

  char* ws = (char*)d_ws;
  size_t off = 0;
  auto alloc = [&](size_t bytes) -> void* {
    off = (off + 255) & ~(size_t)255;
    void* p = ws + off;
    off += bytes;
    return p;
  };
  float*  Qf  = (float*)alloc((size_t)M * (kNH * kHD) * 4);
  float*  Kf  = (float*)alloc((size_t)M * (kNKV * kHD) * 4);
  float*  Vf  = (float*)alloc((size_t)M * (kNKV * kHD) * 4);
  __bf16* Qh  = (__bf16*)alloc((size_t)M * (kNH * kHD) * 2);
  __bf16* Khb = (__bf16*)alloc((size_t)M * (kNKV * kHD) * 2);
  size_t vtBytes = (size_t)kB * kNKV * kHD * kSP * 2;
  __bf16* Vt  = (__bf16*)alloc(vtBytes);
  float*  AO  = (float*)alloc((size_t)M * (kNH * kHD) * 4);

  // 1) QKV projections (bf16 WMMA GEMMs)
  gemm_bf16_wmma_kernel<<<dim3(kNH * kHD / BN, M / BM), 256, 0, stream>>>(
      X, Wq, Qf, M, kNH * kHD, kH);
  gemm_bf16_wmma_kernel<<<dim3(kNKV * kHD / BN, M / BM), 256, 0, stream>>>(
      X, Wk, Kf, M, kNKV * kHD, kH);
  gemm_bf16_wmma_kernel<<<dim3(kNKV * kHD / BN, M / BM), 256, 0, stream>>>(
      X, Wv, Vf, M, kNKV * kHD, kH);

  // 2) zero padded V^T (keeps PV tail tiles finite), then norm + rope
  hipMemsetAsync(Vt, 0, vtBytes, stream);
  norm_rope_kernel<<<M, 256, 0, stream>>>(Qf, Kf, Vf, qw, kw, Qh, Khb, Vt);

  // 3) flash attention: 2*16*128 = 4096 wave-tiles, 4 waves / block
  flash_wmma_kernel<<<(kB * kNH * (kS / 16)) / 4, 128, 0, stream>>>(
      Qh, Khb, Vt, AO);

  // 4) output projection
  gemm_bf16_wmma_kernel<<<dim3(kH / BN, M / BM), 256, 0, stream>>>(
      AO, Wo, out, M, kH, kH);
}
